// StateAlignBlock_33011118637111
// MI455X (gfx1250) — matrix operand order repaired and checked by reading, not compiled, not run on hardware
//
#include <hip/hip_runtime.h>
#include <hip/hip_bf16.h>

// ---------------- problem constants ----------------
#define B_    2
#define C_    128
#define HW_   192
#define NPIX  (HW_*HW_)      // 36864
#define WIN_  4
#define HG_   48
#define NT    2304           // tokens
#define D2_   256            // q/k dim
#define VD_   2048           // v dim
#define NFC_  2
#define NF_   3
#define TOPK_ 5
#define LOCAL_ 4

typedef __attribute__((ext_vector_type(16))) __bf16 v16bf;
typedef __attribute__((ext_vector_type(8)))  float  v8f;

// ---------------- helpers ----------------
__device__ __forceinline__ unsigned short f2bf(float f) {
  unsigned int u = __float_as_uint(f);
  unsigned int r = (u + 0x7FFFu + ((u >> 16) & 1u)) >> 16;   // RNE
  return (unsigned short)r;
}

// A fragment (16x32, M x K), row-major source, ld in elements.
// lane L: M = L&15 ; elements i -> K = 16*(L>>4) + i
__device__ __forceinline__ v16bf load_a(const unsigned short* base, int ld, int lane) {
  const unsigned short* p = base + (size_t)(lane & 15) * ld + ((lane >> 4) << 4);
  union { uint4 u[2]; v16bf v; } c;
  c.u[0] = *(const uint4*)p;
  c.u[1] = *(const uint4*)(p + 8);
  return c.v;
}

// B fragment (32x16, K x N) built from a source stored row-per-N, contiguous in K.
// lane L: N = L&15 ; elements i -> K = 16*(L>>4) + i
__device__ __forceinline__ v16bf load_b(const unsigned short* base, int ld, int lane) {
  const unsigned short* p = base + (size_t)(lane & 15) * ld + ((lane >> 4) << 4);
  union { uint4 u[2]; v16bf v; } c;
  c.u[0] = *(const uint4*)p;
  c.u[1] = *(const uint4*)(p + 8);
  return c.v;
}

// C/D 16x16 f32: lane L holds N = L&15 ; VGPR r holds M = 8*(L>>4)+r
__device__ __forceinline__ void store_c(float* dst, int ld, int lane, v8f acc, float scale) {
  int n = lane & 15, m0 = (lane >> 4) << 3;
#pragma unroll
  for (int r = 0; r < 8; ++r) dst[(size_t)(m0 + r) * ld + n] = acc[r] * scale;
}

__device__ __forceinline__ void ins5(float v, float& t0, float& t1, float& t2, float& t3, float& t4) {
  if (v <= t4) return;
  t4 = v;
  if (t4 > t3) { float t = t3; t3 = t4; t4 = t; }
  if (t3 > t2) { float t = t2; t2 = t3; t3 = t; }
  if (t2 > t1) { float t = t1; t1 = t2; t2 = t; }
  if (t1 > t0) { float t = t0; t0 = t1; t1 = t; }
}

// ---------------- conv kernels (VALU) ----------------
__global__ void conv1x1_kernel(const float* __restrict__ in, const float* __restrict__ w,
                               const float* __restrict__ bias, float* __restrict__ out,
                               int Cin, long inImgStride, long outImgStride) {
  int p4 = blockIdx.x * blockDim.x + threadIdx.x;      // float4 pixel index
  if (p4 >= NPIX / 4) return;
  int oc = blockIdx.y, img = blockIdx.z;
  const float4* inp = (const float4*)(in + (size_t)img * inImgStride);
  const float*  wr  = w + (size_t)oc * Cin;
  float b = bias[oc];
  float4 acc = make_float4(b, b, b, b);
  for (int ic = 0; ic < Cin; ++ic) {
    float4 xv = inp[(size_t)ic * (NPIX / 4) + p4];
    float wv = wr[ic];
    acc.x += wv * xv.x; acc.y += wv * xv.y; acc.z += wv * xv.z; acc.w += wv * xv.w;
  }
  ((float4*)(out + (size_t)img * outImgStride))[(size_t)oc * (NPIX / 4) + p4] = acc;
}

__global__ void dw3x3_kernel(const float* __restrict__ in, const float* __restrict__ w,
                             const float* __restrict__ bias, float* __restrict__ out, int Ctot) {
  int x = blockIdx.x * blockDim.x + threadIdx.x;
  int y = blockIdx.y;
  int nc = blockIdx.z;                // img*Ctot + c
  if (x >= HW_) return;
  int c = nc % Ctot;
  const float* src = in + (size_t)nc * NPIX;
  const float* wr = w + c * 9;
  float acc = bias[c];
#pragma unroll
  for (int dy = 0; dy < 3; ++dy) {
    int yy = y + dy - 1;
    if (yy < 0 || yy >= HW_) continue;
#pragma unroll
    for (int dx = 0; dx < 3; ++dx) {
      int xx = x + dx - 1;
      if (xx < 0 || xx >= HW_) continue;
      acc += wr[dy * 3 + dx] * src[yy * HW_ + xx];
    }
  }
  out[(size_t)nc * NPIX + (size_t)y * HW_ + x] = acc;
}

// depthwise 4x4, stride 4, pad 1  -> 48x48
__global__ void dwwin_kernel(const float* __restrict__ in, const float* __restrict__ w,
                             const float* __restrict__ bias, float* __restrict__ out, int Ctot) {
  int ox = blockIdx.x * blockDim.x + threadIdx.x;
  int oy = blockIdx.y;
  int nc = blockIdx.z;
  if (ox >= HG_) return;
  int c = nc % Ctot;
  const float* src = in + (size_t)nc * NPIX;
  const float* wr = w + c * 16;
  float acc = bias[c];
#pragma unroll
  for (int ky = 0; ky < 4; ++ky) {
    int yy = oy * 4 - 1 + ky;
    if (yy < 0 || yy >= HW_) continue;
#pragma unroll
    for (int kx = 0; kx < 4; ++kx) {
      int xx = ox * 4 - 1 + kx;
      if (xx < 0 || xx >= HW_) continue;
      acc += wr[ky * 4 + kx] * src[yy * HW_ + xx];
    }
  }
  out[(size_t)nc * (HG_ * HG_) + (size_t)oy * HG_ + ox] = acc;
}

// ---------------- packing / normalization ----------------
// src: [B][256][48][48] channel-major. Writes l2-normalized token rows:
// bf16 at dst_bf + b*bfBStride + tok*256 ; optional fp32 at dst_f32 + b*f32BStride + tok*256
__global__ void norm_pack_kernel(const float* __restrict__ src,
                                 unsigned short* __restrict__ dst_bf, long bfBStride,
                                 float* __restrict__ dst_f32, long f32BStride) {
  int tok = blockIdx.x, b = blockIdx.y, tid = threadIdx.x;  // 64 threads
  __shared__ float red[64];
  float v[4]; float ss = 0.f;
#pragma unroll
  for (int i = 0; i < 4; ++i) {
    int ch = tid + i * 64;
    v[i] = src[((size_t)b * D2_ + ch) * (HG_ * HG_) + tok];
    ss += v[i] * v[i];
  }
  red[tid] = ss; __syncthreads();
  for (int s = 32; s > 0; s >>= 1) { if (tid < s) red[tid] += red[tid + s]; __syncthreads(); }
  float scale = 1.f / fmaxf(sqrtf(red[0]), 1e-12f);
  unsigned short* db = dst_bf + (size_t)b * bfBStride + (size_t)tok * D2_;
  float* df = dst_f32 ? dst_f32 + (size_t)b * f32BStride + (size_t)tok * D2_ : (float*)0;
#pragma unroll
  for (int i = 0; i < 4; ++i) {
    int ch = tid + i * 64;
    float nv = v[i] * scale;
    db[ch] = f2bf(nv);
    if (df) df[ch] = nv;
  }
}

__global__ void kcache_kernel(const float* __restrict__ kc, unsigned short* __restrict__ kall) {
  size_t i = (size_t)blockIdx.x * blockDim.x + threadIdx.x;
  size_t total = (size_t)B_ * NFC_ * NT * D2_;
  if (i >= total) return;
  int d = (int)(i % D2_); size_t t = i / D2_;
  int tok = (int)(t % NT); t /= NT;
  int f = (int)(t % NFC_); int b = (int)(t / NFC_);
  kall[(((size_t)b * NF_ + f) * NT + tok) * D2_ + d] = f2bf(kc[i]);
}

// v_cached [B][2][NT][VD] fp32 -> vT bf16 [B][3][VD][NT] frames 0,1
__global__ void vcacheT_kernel(const float* __restrict__ vc, unsigned short* __restrict__ vT) {
  size_t i = (size_t)blockIdx.x * blockDim.x + threadIdx.x;
  size_t total = (size_t)B_ * NFC_ * VD_ * NT;
  if (i >= total) return;
  int tok = (int)(i % NT); size_t t = i / NT;
  int feat = (int)(t % VD_); t /= VD_;
  int f = (int)(t % NFC_); int b = (int)(t / NFC_);
  float val = vc[(((size_t)b * NFC_ + f) * NT + tok) * VD_ + feat];
  vT[(((size_t)b * NF_ + f) * VD_ + feat) * NT + tok] = f2bf(val);
}

// v_full [B][128][192][192] -> v_new token layout: fp32 to out_v frame 1, bf16 to vT frame 2
__global__ void vnew_kernel(const float* __restrict__ vfull, float* __restrict__ vout,
                            unsigned short* __restrict__ vT) {
  size_t i = (size_t)blockIdx.x * blockDim.x + threadIdx.x;
  if (i >= (size_t)B_ * NT * VD_) return;
  int feat = (int)(i % VD_); size_t t = i / VD_;
  int tok = (int)(t % NT); int b = (int)(t / NT);
  int ch = feat % C_, pp = feat / C_, p2 = pp % WIN_, p1 = pp / WIN_;
  int y = tok / HG_, x = tok % HG_;
  float val = vfull[(((size_t)b * C_ + ch) * HW_ + (p1 * HG_ + y)) * HW_ + (p2 * HG_ + x)];
  vout[(((size_t)b * NFC_ + 1) * NT + tok) * VD_ + feat] = val;
  vT[(((size_t)b * NF_ + 2) * VD_ + feat) * NT + tok] = f2bf(val);
}

// ---------------- WMMA GEMMs ----------------
// S[2304,2304] = temperature * q[2304,256] . k[2304,256]^T   (bf16 in, f32 out)
// block = 128 threads (4 waves); wave tile 32x64; block tile 128x64; grid (36, 18)
__global__ void __launch_bounds__(128) gemm_qk_kernel(const unsigned short* __restrict__ q,
                                                      const unsigned short* __restrict__ k,
                                                      float* __restrict__ S,
                                                      const float* __restrict__ temp) {
  int lane = threadIdx.x & 31, wave = threadIdx.x >> 5;
  int row0 = blockIdx.y * 128 + wave * 32;
  int col0 = blockIdx.x * 64;
  v8f acc[2][4] = {};
  for (int kk = 0; kk < D2_; kk += 32) {
    v16bf a0 = load_a(q + (size_t)row0 * D2_ + kk, D2_, lane);
    v16bf a1 = load_a(q + (size_t)(row0 + 16) * D2_ + kk, D2_, lane);
#pragma unroll
    for (int t = 0; t < 4; ++t) {
      v16bf b = load_b(k + (size_t)(col0 + t * 16) * D2_ + kk, D2_, lane);
      acc[0][t] = __builtin_amdgcn_wmma_f32_16x16x32_bf16(false, a0, false, b,
                                                          (short)0, acc[0][t], false, false);
      acc[1][t] = __builtin_amdgcn_wmma_f32_16x16x32_bf16(false, a1, false, b,
                                                          (short)0, acc[1][t], false, false);
    }
  }
  float ts = temp[0];
#pragma unroll
  for (int r = 0; r < 2; ++r)
#pragma unroll
    for (int t = 0; t < 4; ++t)
      store_c(S + (size_t)(row0 + r * 16) * NT + col0 + t * 16, NT, lane, acc[r][t], ts);
}

// O[2304,2048] = A[2304,2304](bf16) . V[2304,2048]  via vT[2048,2304] bf16
// block = 256 threads (8 waves); wave tile 32x64; block tile 256x64; grid (32, 9)
__global__ void __launch_bounds__(256) gemm_av_kernel(const unsigned short* __restrict__ A,
                                                      const unsigned short* __restrict__ vT,
                                                      float* __restrict__ O) {
  int lane = threadIdx.x & 31, wave = threadIdx.x >> 5;
  int row0 = blockIdx.y * 256 + wave * 32;
  int col0 = blockIdx.x * 64;
  v8f acc[2][4] = {};
  for (int kk = 0; kk < NT; kk += 32) {
    v16bf a0 = load_a(A + (size_t)row0 * NT + kk, NT, lane);
    v16bf a1 = load_a(A + (size_t)(row0 + 16) * NT + kk, NT, lane);
#pragma unroll
    for (int t = 0; t < 4; ++t) {
      v16bf b = load_b(vT + (size_t)(col0 + t * 16) * NT + kk, NT, lane);
      acc[0][t] = __builtin_amdgcn_wmma_f32_16x16x32_bf16(false, a0, false, b,
                                                          (short)0, acc[0][t], false, false);
      acc[1][t] = __builtin_amdgcn_wmma_f32_16x16x32_bf16(false, a1, false, b,
                                                          (short)0, acc[1][t], false, false);
    }
  }
#pragma unroll
  for (int r = 0; r < 2; ++r)
#pragma unroll
    for (int t = 0; t < 4; ++t)
      store_c(O + (size_t)(row0 + r * 16) * VD_ + col0 + t * 16, VD_, lane, acc[r][t], 1.0f);
}

// ---------------- top-k + local mask + clipped softmax (per row) ----------------
__global__ void __launch_bounds__(128) attn_mask_softmax_kernel(const float* __restrict__ S,
                                                                unsigned short* __restrict__ A) {
  __shared__ float row[NT];
  __shared__ float red[128];
  __shared__ float tops[128 * 5];
  __shared__ float sh_kth;
  int tid = threadIdx.x, r = blockIdx.x;
  const float* src = S + (size_t)r * NT;
  for (int j = tid; j < NT; j += 128) row[j] = src[j];
  __syncthreads();
  const float NEG = -3.0e38f;
  float t0 = NEG, t1 = NEG, t2 = NEG, t3 = NEG, t4 = NEG;
  for (int j = tid; j < NT; j += 128) ins5(row[j], t0, t1, t2, t3, t4);
  tops[tid * 5 + 0] = t0; tops[tid * 5 + 1] = t1; tops[tid * 5 + 2] = t2;
  tops[tid * 5 + 3] = t3; tops[tid * 5 + 4] = t4;
  __syncthreads();
  if (tid == 0) {
    float m0 = NEG, m1 = NEG, m2 = NEG, m3 = NEG, m4 = NEG;
    for (int i = 0; i < 128 * 5; ++i) ins5(tops[i], m0, m1, m2, m3, m4);
    sh_kth = m4;
  }
  __syncthreads();
  float kth = sh_kth;
  int ry = r / HG_, rx = r % HG_;
  float lmax = NEG;
  for (int j = tid; j < NT; j += 128) {
    float s = row[j];
    int y = j / HG_, x = j % HG_;
    int dist = (ry > y ? ry - y : y - ry) + (rx > x ? rx - x : x - rx);
    float m = (s >= kth ? s : 0.f) + (dist <= LOCAL_ ? s : 0.f);
    row[j] = m;
    if (m != 0.f) lmax = fmaxf(lmax, m);
  }
  red[tid] = lmax; __syncthreads();
  for (int s2 = 64; s2 > 0; s2 >>= 1) { if (tid < s2) red[tid] = fmaxf(red[tid], red[tid + s2]); __syncthreads(); }
  float mx = red[0];
  __syncthreads();
  float lsum = 0.f;
  for (int j = tid; j < NT; j += 128) { float m = row[j]; if (m != 0.f) lsum += __expf(m - mx); }
  red[tid] = lsum; __syncthreads();
  for (int s2 = 64; s2 > 0; s2 >>= 1) { if (tid < s2) red[tid] += red[tid + s2]; __syncthreads(); }
  float inv = 1.f / red[0];
  unsigned short* dst = A + (size_t)r * NT;
  for (int j = tid; j < NT; j += 128) {
    float m = row[j];
    dst[j] = f2bf(m != 0.f ? __expf(m - mx) * inv : 0.f);
  }
}

// ---------------- unpack + proj 1x1 conv (per slice) ----------------
__global__ void proj_kernel(const float* __restrict__ O, const float* __restrict__ w,
                            const float* __restrict__ bias, float* __restrict__ out) {
  int x = blockIdx.x * blockDim.x + threadIdx.x;
  int h = blockIdx.y;
  int oc = blockIdx.z;
  if (x >= HW_) return;
  int p1 = h / HG_, y = h % HG_, p2 = x / HG_, xx = x % HG_;
  int tok = y * HG_ + xx;
  const float* orow = O + (size_t)tok * VD_ + (size_t)(p1 * WIN_ + p2) * C_;
  const float* wr = w + (size_t)oc * C_;
  float acc = bias[oc];
#pragma unroll 8
  for (int ic = 0; ic < C_; ++ic) acc += wr[ic] * orow[ic];
  out[((size_t)oc * HW_ + h) * HW_ + x] = acc;
}

// ---------------- workspace layout (bytes) ----------------
static const size_t WS_TMPA  = 0;                                  // [B][256][192][192] f32
static const size_t WS_QK    = WS_TMPA  + (size_t)B_*2*C_*NPIX*4;  // [B][256][192][192] f32
static const size_t WS_TMPB  = WS_QK    + (size_t)B_*2*C_*NPIX*4;  // [B][128][192][192] f32
static const size_t WS_VFULL = WS_TMPB  + (size_t)B_*C_*NPIX*4;    // [B][128][192][192] f32
static const size_t WS_KDS   = WS_VFULL + (size_t)B_*C_*NPIX*4;    // [B][256][48][48] f32
static const size_t WS_QDS   = WS_KDS   + (size_t)B_*D2_*HG_*HG_*4;   // [B][256][48][48] f32
static const size_t WS_QBF   = WS_QDS   + (size_t)B_*D2_*HG_*HG_*4;   // [B][NT][256] bf16
static const size_t WS_KALL  = WS_QBF   + (size_t)B_*NT*D2_*2;        // [B][3][NT][256] bf16
static const size_t WS_VT    = WS_KALL  + (size_t)B_*NF_*NT*D2_*2;    // [B][3][VD][NT] bf16
static const size_t WS_S     = WS_VT    + (size_t)B_*NF_*VD_*NT*2;    // [NT][NT] f32 (per slice)
static const size_t WS_A     = WS_S     + (size_t)NT*NT*4;            // [NT][NT] bf16
static const size_t WS_O     = WS_A     + (size_t)NT*NT*2;            // [NT][VD] f32

extern "C" void kernel_launch(void* const* d_in, const int* in_sizes, int n_in,
                              void* d_out, int out_size, void* d_ws, size_t ws_size,
                              hipStream_t stream) {
  const float* x        = (const float*)d_in[0];
  const float* k_cached = (const float*)d_in[1];
  const float* v_cached = (const float*)d_in[2];
  const float* temp     = (const float*)d_in[3];
  const float* qk_w = (const float*)d_in[4],  *qk_b = (const float*)d_in[5];
  const float* qk_dw_w = (const float*)d_in[6], *qk_dw_b = (const float*)d_in[7];
  const float* v_w = (const float*)d_in[8],   *v_b = (const float*)d_in[9];
  const float* v_dw_w = (const float*)d_in[10], *v_dw_b = (const float*)d_in[11];
  const float* k2_w = (const float*)d_in[12], *k2_b = (const float*)d_in[13];
  const float* k2_dw_w = (const float*)d_in[14], *k2_dw_b = (const float*)d_in[15];
  const float* q2_w = (const float*)d_in[16], *q2_b = (const float*)d_in[17];
  const float* q2_dw_w = (const float*)d_in[18], *q2_dw_b = (const float*)d_in[19];
  const float* proj_w = (const float*)d_in[20], *proj_b = (const float*)d_in[21];

  char* ws = (char*)d_ws;
  float* tmpa  = (float*)(ws + WS_TMPA);
  float* qkf   = (float*)(ws + WS_QK);
  float* tmpb  = (float*)(ws + WS_TMPB);
  float* vfull = (float*)(ws + WS_VFULL);
  float* kds   = (float*)(ws + WS_KDS);
  float* qds   = (float*)(ws + WS_QDS);
  unsigned short* qbf   = (unsigned short*)(ws + WS_QBF);
  unsigned short* kall  = (unsigned short*)(ws + WS_KALL);
  unsigned short* vtbf  = (unsigned short*)(ws + WS_VT);
  float* Sbuf = (float*)(ws + WS_S);
  unsigned short* Abuf = (unsigned short*)(ws + WS_A);
  float* Obuf = (float*)(ws + WS_O);

  float* out   = (float*)d_out;
  float* out_k = out + (size_t)B_ * NF_ * C_ * NPIX;        // 28,311,552
  float* out_v = out_k + (size_t)B_ * NFC_ * NT * D2_;      // +2,359,296

  dim3 cgrd((NPIX / 4 + 255) / 256, 2 * C_, B_);
  dim3 cgrd1((NPIX / 4 + 255) / 256, C_, B_);
  dim3 dwg((HW_ + 63) / 64, HW_, B_ * 2 * C_);
  dim3 dwg1((HW_ + 63) / 64, HW_, B_ * C_);
  dim3 wing(1, HG_, B_ * 2 * C_);

  // qk path
  conv1x1_kernel<<<cgrd, 256, 0, stream>>>(x, qk_w, qk_b, tmpa, C_, (long)C_*NPIX, (long)2*C_*NPIX);
  dw3x3_kernel<<<dwg, 64, 0, stream>>>(tmpa, qk_dw_w, qk_dw_b, qkf, 2 * C_);
  // v path
  conv1x1_kernel<<<cgrd1, 256, 0, stream>>>(x, v_w, v_b, tmpb, C_, (long)C_*NPIX, (long)C_*NPIX);
  dw3x3_kernel<<<dwg1, 64, 0, stream>>>(tmpb, v_dw_w, v_dw_b, vfull, C_);
  // k2 path (input = channels 128..255 of qkf)
  conv1x1_kernel<<<cgrd, 256, 0, stream>>>(qkf + (size_t)C_*NPIX, k2_w, k2_b, tmpa, C_, (long)2*C_*NPIX, (long)2*C_*NPIX);
  dwwin_kernel<<<wing, 64, 0, stream>>>(tmpa, k2_dw_w, k2_dw_b, kds, 2 * C_);
  // q2 path (input = channels 0..127 of qkf)
  conv1x1_kernel<<<cgrd, 256, 0, stream>>>(qkf, q2_w, q2_b, tmpa, C_, (long)2*C_*NPIX, (long)2*C_*NPIX);
  dwwin_kernel<<<wing, 64, 0, stream>>>(tmpa, q2_dw_w, q2_dw_b, qds, 2 * C_);

  // normalize + pack: q -> qbf ; k -> kall frame 2 (bf16) + out_k frame 1 (fp32)
  norm_pack_kernel<<<dim3(NT, B_), 64, 0, stream>>>(qds, qbf, (long)NT * D2_, (float*)0, 0);
  norm_pack_kernel<<<dim3(NT, B_), 64, 0, stream>>>(kds, kall + (size_t)2 * NT * D2_,
                                                   (long)NF_ * NT * D2_,
                                                   out_k + (size_t)NT * D2_,
                                                   (long)NFC_ * NT * D2_);
  // caches
  {
    size_t n1 = (size_t)B_ * NFC_ * NT * D2_;
    kcache_kernel<<<(unsigned)((n1 + 255) / 256), 256, 0, stream>>>(k_cached, kall);
    size_t n2 = (size_t)B_ * NFC_ * VD_ * NT;
    vcacheT_kernel<<<(unsigned)((n2 + 255) / 256), 256, 0, stream>>>(v_cached, vtbf);
    size_t n3 = (size_t)B_ * NT * VD_;
    vnew_kernel<<<(unsigned)((n3 + 255) / 256), 256, 0, stream>>>(vfull, out_v, vtbf);
  }
  // fp32 cache pass-through outputs (frame 0 = last cached frame)
  for (int b = 0; b < B_; ++b) {
    hipMemcpyAsync(out_k + (size_t)(b * NFC_) * NT * D2_,
                   k_cached + (size_t)(b * NFC_ + 1) * NT * D2_,
                   (size_t)NT * D2_ * sizeof(float), hipMemcpyDeviceToDevice, stream);
    hipMemcpyAsync(out_v + (size_t)(b * NFC_) * NT * VD_,
                   v_cached + (size_t)(b * NFC_ + 1) * NT * VD_,
                   (size_t)NT * VD_ * sizeof(float), hipMemcpyDeviceToDevice, stream);
  }

  // attention per (b, frame) slice
  for (int b = 0; b < B_; ++b) {
    for (int f = 0; f < NF_; ++f) {
      const unsigned short* qb = qbf + (size_t)b * NT * D2_;
      const unsigned short* kb = kall + ((size_t)b * NF_ + f) * NT * D2_;
      const unsigned short* vb = vtbf + ((size_t)b * NF_ + f) * VD_ * NT;
      gemm_qk_kernel<<<dim3(NT / 64, NT / 128), 128, 0, stream>>>(qb, kb, Sbuf, temp);
      attn_mask_softmax_kernel<<<NT, 128, 0, stream>>>(Sbuf, Abuf);
      gemm_av_kernel<<<dim3(VD_ / 64, NT / 256), 256, 0, stream>>>(Abuf, vb, Obuf);
      proj_kernel<<<dim3((HW_ + 63) / 64, HW_, C_), 64, 0, stream>>>(
          Obuf, proj_w, proj_b, out + ((size_t)b * NF_ + f) * C_ * NPIX);
    }
  }
}
